// Attention_1477468750217
// MI455X (gfx1250) — matrix operand order repaired and checked by reading, not compiled, not run on hardware
//
#include <hip/hip_runtime.h>

// ---------------------------------------------------------------------------
// MI455X (gfx1250) attention: bf16 WMMA GEMMs + RoPE + flash-attention.
// Roofline: ~0.6 TFLOP, ~0.5 GB mandatory traffic @ 23.3 TB/s -> run all
// matmuls on v_wmma_f32_16x16x32_bf16, keep intermediates bf16, and hide
// global-load latency with double-buffered LDS staging.
// ---------------------------------------------------------------------------

typedef unsigned short u16;
typedef unsigned int u32;
typedef __bf16 v16bf __attribute__((ext_vector_type(16)));
typedef float v8f __attribute__((ext_vector_type(8)));
typedef u32 u32x4 __attribute__((ext_vector_type(4)));

#define HIDDEN 4096
#define SLEN 2048
#define BATCH 2
#define NHEADS 32
#define KVHEADS 8
#define HDIM 128
#define MTOT (BATCH * SLEN) /* 4096 */
#define ATTN_SCALE 0.08838834764831845f /* 1/sqrt(128) */

// padded LDS row strides (u16 units): +8 u16 (16B) kills the power-of-two
// bank pattern; 40*2=80B stride -> 16-lane fragment gathers hit 16 disjoint
// 4-bank groups (conflict-free), and rows stay 16B aligned for ds_load_b128.
#define LDP32 40   /* for 32-wide tiles  */
#define LDP128 136 /* for 128-wide tiles */

__device__ __forceinline__ u16 f2bf(float f) {
  u32 x = __float_as_uint(f);
  x += 0x7FFFu + ((x >> 16) & 1u); // round-to-nearest-even
  return (u16)(x >> 16);
}
__device__ __forceinline__ float bf2f(u16 h) {
  return __uint_as_float(((u32)h) << 16);
}

union FragBF {
  v16bf v;
  u32 u[8];
  u32x4 q[2];
};

// 16x32 bf16 A fragment from row-major [16][ld] source.
// Lane L: row = L&15; lanes 0-15 hold K={0..7,16..23}, lanes 16-31 {8..15,24..31}
// (CDNA5 ISA 7.12.2 16-bit A-matrix layout). Two 16B loads per lane.
__device__ __forceinline__ void load_frag_a(const u16* src, int ld, FragBF& f, int lane) {
  const u16* p = src + (lane & 15) * ld + ((lane >> 4) << 3);
  f.q[0] = *(const u32x4*)(p);
  f.q[1] = *(const u32x4*)(p + 16);
}
// 32x16 bf16 B fragment where element (k,n) lives at src[n*ld + k].
// Lane L: n = L&15; lanes 0-15 hold K={0..7,16..23}, lanes 16-31 {8..15,24..31}, packed as K-pairs per dword.
__device__ __forceinline__ void load_frag_b(const u16* src, int ld, FragBF& f, int lane) {
  const u16* p = src + (lane & 15) * ld + ((lane >> 4) << 3);
  f.q[0] = *(const u32x4*)(p);
  f.q[1] = *(const u32x4*)(p + 16);
}
__device__ __forceinline__ v8f wmma_bf16(const FragBF& a, const FragBF& b, v8f c) {
  return __builtin_amdgcn_wmma_f32_16x16x32_bf16(false, a.v, false, b.v,
                                                 (short)0, c, false, false);
}

// ---------------------------------------------------------------------------
__global__ __launch_bounds__(256)
void cast_f32_bf16(const float* __restrict__ in, u16* __restrict__ out, int n) {
  int i = blockIdx.x * 256 + threadIdx.x;
  int stride = gridDim.x * 256;
  for (; i < n; i += stride) out[i] = f2bf(in[i]);
}

// ---------------------------------------------------------------------------
// C[M=4096, N] = A[M,4096] x W[N,4096]^T.
// Block tile 128x128, 8 waves, each wave 32x64 = 2x4 WMMA tiles, K-step 32.
// Double-buffered LDS: next tile's global_load_b128s are issued before the
// WMMA phase so s_wait_loadcnt lands after the math (latency hidden).
// mode 0: store bf16 head-major [B][heads][S][D];  mode 1: store f32 [M][N].
__global__ __launch_bounds__(256)
void gemm_bf16_kernel(const u16* __restrict__ A, const u16* __restrict__ W,
                      void* __restrict__ out, int N, int mode, int heads) {
  __shared__ __align__(16) u16 As[2][128 * LDP32];
  __shared__ __align__(16) u16 Bs[2][128 * LDP32];
  const int t = threadIdx.x;
  const int lane = t & 31;
  const int wid = t >> 5;
  const int wm = wid >> 1; // 0..3 -> M offset wm*32
  const int wn = wid & 1;  // 0..1 -> N offset wn*64
  const int n0 = blockIdx.x * 128;
  const int m0 = blockIdx.y * 128;

  // staging: each thread moves 2x16B of A and 2x16B of B per K-step
  const int sr = t >> 2;        // 0..63 (+64 on second iter)
  const int sc = (t & 3) << 3;  // 0,8,16,24 (u16 units)

  u32x4 ra[2], rb[2];
#pragma unroll
  for (int i = 0; i < 2; ++i) {
    int r = sr + i * 64;
    ra[i] = *(const u32x4*)&A[(size_t)(m0 + r) * HIDDEN + sc];
    rb[i] = *(const u32x4*)&W[(size_t)(n0 + r) * HIDDEN + sc];
  }
#pragma unroll
  for (int i = 0; i < 2; ++i) {
    int r = sr + i * 64;
    *(u32x4*)&As[0][r * LDP32 + sc] = ra[i];
    *(u32x4*)&Bs[0][r * LDP32 + sc] = rb[i];
  }

  v8f acc[2][4] = {};
  const int NK = HIDDEN / 32;
  for (int kt = 0; kt < NK; ++kt) {
    const int buf = kt & 1;
    __syncthreads(); // buf stores visible; prior reads of buf^1 done
    if (kt + 1 < NK) { // prefetch next K-tile into registers (overlaps WMMA)
      const int k0 = (kt + 1) * 32;
#pragma unroll
      for (int i = 0; i < 2; ++i) {
        int r = sr + i * 64;
        ra[i] = *(const u32x4*)&A[(size_t)(m0 + r) * HIDDEN + k0 + sc];
        rb[i] = *(const u32x4*)&W[(size_t)(n0 + r) * HIDDEN + k0 + sc];
      }
    }
    FragBF af[2], bfm[4];
#pragma unroll
    for (int mt = 0; mt < 2; ++mt)
      load_frag_a(&As[buf][(wm * 32 + mt * 16) * LDP32], LDP32, af[mt], lane);
#pragma unroll
    for (int nt = 0; nt < 4; ++nt)
      load_frag_b(&Bs[buf][(wn * 64 + nt * 16) * LDP32], LDP32, bfm[nt], lane);
#pragma unroll
    for (int mt = 0; mt < 2; ++mt)
#pragma unroll
      for (int nt = 0; nt < 4; ++nt)
        acc[mt][nt] = wmma_bf16(af[mt], bfm[nt], acc[mt][nt]);
    if (kt + 1 < NK) { // fill the other buffer (guarded by next top barrier)
#pragma unroll
      for (int i = 0; i < 2; ++i) {
        int r = sr + i * 64;
        *(u32x4*)&As[buf ^ 1][r * LDP32 + sc] = ra[i];
        *(u32x4*)&Bs[buf ^ 1][r * LDP32 + sc] = rb[i];
      }
    }
  }

  // C layout: lane holds n = lane&15, row m = (lane>>4)*8 + v.
  const int halfb = (lane >> 4) << 3;
  const int ncol = lane & 15;
#pragma unroll
  for (int mt = 0; mt < 2; ++mt)
#pragma unroll
    for (int nt = 0; nt < 4; ++nt)
#pragma unroll
      for (int v = 0; v < 8; ++v) {
        int m = m0 + wm * 32 + mt * 16 + halfb + v;
        int n = n0 + wn * 64 + nt * 16 + ncol;
        float val = acc[mt][nt][v];
        if (mode == 0) {
          int b = m >> 11, s = m & (SLEN - 1);
          int h = n >> 7, d = n & (HDIM - 1);
          ((u16*)out)[(((size_t)(b * heads + h)) * SLEN + s) * HDIM + d] = f2bf(val);
        } else {
          ((float*)out)[(size_t)m * N + n] = val;
        }
      }
}

// ---------------------------------------------------------------------------
// RoPE in place on head-major bf16 [B][heads][S][128]; pairs (d, d+64).
// grid = heads*1024 blocks of 256 (exact cover).
__global__ __launch_bounds__(256)
void rope_kernel(u16* __restrict__ X, const int* __restrict__ pos, int heads) {
  int i = blockIdx.x * 256 + threadIdx.x;
  int d = i & 63;
  int s = (i >> 6) & (SLEN - 1);
  int bh = i >> 17; // 0 .. 2*heads-1
  size_t base = ((size_t)bh * SLEN + s) * HDIM;
  float p = (float)pos[s];
  // inv_freq = 10000^(-d/64) = exp(-d * ln(10000)/64)
  float inv = __expf(-(float)d * (9.210340371976184f / 64.0f));
  float fr = p * inv;
  float sn, cs;
  __sincosf(fr, &sn, &cs);
  float x1 = bf2f(X[base + d]);
  float x2 = bf2f(X[base + d + 64]);
  X[base + d] = f2bf(x1 * cs - x2 * sn);
  X[base + d + 64] = f2bf(x2 * cs + x1 * sn);
}

// ---------------------------------------------------------------------------
// Flash attention: block = (b, head, 64 q-rows); 4 waves x 16 q-rows.
// KV tile = 32 rows staged to LDS (K row-major padded, V transposed padded).
__global__ __launch_bounds__(128)
void attn_kernel(const u16* __restrict__ Q, const u16* __restrict__ K,
                 const u16* __restrict__ V, u16* __restrict__ AO) {
  __shared__ __align__(16) u16 Ks[32 * LDP128];   // [kv][d], padded rows
  __shared__ __align__(16) u16 Vs[128 * LDP32];   // transposed: [d][kv]
  __shared__ __align__(16) u16 Ps[4][16 * LDP32]; // per-wave P relayout buffer

  const int t = threadIdx.x;
  const int lane = t & 31;
  const int w = t >> 5;
  const int bid = blockIdx.x;
  const int qb = bid & 31;  // q-block (64 rows each)
  const int bh = bid >> 5;  // 0..63
  const int b = bh >> 5;
  const int h = bh & 31;
  const int kvh = h >> 2;   // GQA: kv head = h / GROUPS (consecutive repeat)

  const int q0 = qb * 64;
  const int qw0 = q0 + w * 16;
  const int halfb = (lane >> 4) << 3;
  const int ncol = lane & 15;

  const u16* Qbase = Q + (((size_t)b * NHEADS + h) * SLEN + qw0) * HDIM;
  const u16* Kbase = K + ((size_t)b * KVHEADS + kvh) * SLEN * HDIM;
  const u16* Vbase = V + ((size_t)b * KVHEADS + kvh) * SLEN * HDIM;

  FragBF qf[4]; // 16 q-rows x 128 dims, preloaded once
#pragma unroll
  for (int c = 0; c < 4; ++c) load_frag_a(Qbase + c * 32, HDIM, qf[c], lane);

  v8f o[8] = {};
  float Mrow[8], Lrow[8];
#pragma unroll
  for (int v = 0; v < 8; ++v) { Mrow[v] = -1e30f; Lrow[v] = 0.f; }

  const int ktiles = qb * 2 + 2; // cover kv 0 .. q0+63
  for (int kt = 0; kt < ktiles; ++kt) {
    const int kv0 = kt * 32;
    __syncthreads();
#pragma unroll
    for (int i = 0; i < 4; ++i) { // K tile 32x128 bf16 = 512 x 16B
      int id = i * 128 + t;
      int r = id >> 4, c = (id & 15) << 3;
      *(u32x4*)&Ks[r * LDP128 + c] = *(const u32x4*)&Kbase[(size_t)(kv0 + r) * HDIM + c];
    }
#pragma unroll
    for (int i = 0; i < 16; ++i) { // V tile transposed into LDS
      int id = i * 128 + t;
      int r = id >> 6, c2 = (id & 63) << 1;
      u32 pr = *(const u32*)&Vbase[(size_t)(kv0 + r) * HDIM + c2];
      Vs[c2 * LDP32 + r] = (u16)(pr & 0xFFFFu);
      Vs[(c2 + 1) * LDP32 + r] = (u16)(pr >> 16);
    }
    __syncthreads();

    if (kv0 <= qw0 + 15) { // wave-level causal skip (no barriers inside)
      v8f sc[2] = {};
#pragma unroll
      for (int c = 0; c < 4; ++c) { // S(16x32) = Q x K^T, 8 WMMA
        FragBF bk;
        load_frag_b(&Ks[c * 32], LDP128, bk, lane);
        sc[0] = wmma_bf16(qf[c], bk, sc[0]);
        load_frag_b(&Ks[16 * LDP128 + c * 32], LDP128, bk, lane);
        sc[1] = wmma_bf16(qf[c], bk, sc[1]);
      }
      // online softmax; row m lives across 16 lanes of one half-wave
#pragma unroll
      for (int v = 0; v < 8; ++v) {
        int mg = qw0 + halfb + v;
        float s0 = sc[0][v] * ATTN_SCALE;
        float s1 = sc[1][v] * ATTN_SCALE;
        bool k0m = (kv0 + ncol) > mg;
        bool k1m = (kv0 + 16 + ncol) > mg;
        if (k0m) s0 = -1e30f;
        if (k1m) s1 = -1e30f;
        float rmax = fmaxf(s0, s1);
#pragma unroll
        for (int off = 1; off < 16; off <<= 1)
          rmax = fmaxf(rmax, __shfl_xor(rmax, off, 16));
        float newM = fmaxf(Mrow[v], rmax);
        float alpha = __expf(Mrow[v] - newM);
        float p0 = k0m ? 0.f : __expf(s0 - newM);
        float p1 = k1m ? 0.f : __expf(s1 - newM);
        float rsum = p0 + p1;
#pragma unroll
        for (int off = 1; off < 16; off <<= 1)
          rsum += __shfl_xor(rsum, off, 16);
        Lrow[v] = Lrow[v] * alpha + rsum;
        Mrow[v] = newM;
#pragma unroll
        for (int j = 0; j < 8; ++j) o[j][v] *= alpha;
        // C-layout -> A-layout relayout through per-wave LDS
        Ps[w][(halfb + v) * LDP32 + ncol] = f2bf(p0);
        Ps[w][(halfb + v) * LDP32 + 16 + ncol] = f2bf(p1);
      }
      asm volatile("s_wait_dscnt 0" ::: "memory"); // wave-local LDS RAW fence
      FragBF pf;
      load_frag_a(&Ps[w][0], LDP32, pf, lane);
#pragma unroll
      for (int j = 0; j < 8; ++j) { // O(16x128) += P x V, 8 WMMA
        FragBF bv;
        load_frag_b(&Vs[(j * 16) * LDP32], LDP32, bv, lane);
        o[j] = wmma_bf16(pf, bv, o[j]);
      }
    }
  }

  // normalize and write bf16 attn output [b][s][h*128+d]
  u16* dst = AO + (size_t)b * SLEN * (NHEADS * HDIM) + (size_t)h * HDIM;
#pragma unroll
  for (int v = 0; v < 8; ++v) {
    int sg = qw0 + halfb + v;
    float inv = 1.0f / Lrow[v];
#pragma unroll
    for (int j = 0; j < 8; ++j) {
      int d = j * 16 + ncol;
      dst[(size_t)sg * (NHEADS * HDIM) + d] = f2bf(o[j][v] * inv);
    }
  }
}

// ---------------------------------------------------------------------------
extern "C" void kernel_launch(void* const* d_in, const int* in_sizes, int n_in,
                              void* d_out, int out_size, void* d_ws, size_t ws_size,
                              hipStream_t stream) {
  (void)in_sizes; (void)n_in; (void)out_size; (void)ws_size;
  const float* hidden = (const float*)d_in[0];
  const int* pos = (const int*)d_in[1];
  const float* Wq = (const float*)d_in[2];
  const float* Wk = (const float*)d_in[3];
  const float* Wv = (const float*)d_in[4];
  const float* Wo = (const float*)d_in[5];

  // workspace carve-up (bf16), ~192 MiB total
  u16* p = (u16*)d_ws;
  u16* hbf = p; p += (size_t)MTOT * HIDDEN;                     // 16.7M
  u16* Wqb = p; p += (size_t)HIDDEN * HIDDEN;                   // 16.7M
  u16* Wkb = p; p += (size_t)(KVHEADS * HDIM) * HIDDEN;         // 4.2M
  u16* Wvb = p; p += (size_t)(KVHEADS * HDIM) * HIDDEN;         // 4.2M
  u16* Wob = p; p += (size_t)HIDDEN * HIDDEN;                   // 16.7M
  u16* Qh  = p; p += (size_t)BATCH * NHEADS * SLEN * HDIM;      // 16.7M
  u16* Kh  = p; p += (size_t)BATCH * KVHEADS * SLEN * HDIM;     // 4.2M
  u16* Vh  = p; p += (size_t)BATCH * KVHEADS * SLEN * HDIM;     // 4.2M
  u16* AO  = p; p += (size_t)MTOT * HIDDEN;                     // 16.7M

  cast_f32_bf16<<<2048, 256, 0, stream>>>(hidden, hbf, MTOT * HIDDEN);
  cast_f32_bf16<<<2048, 256, 0, stream>>>(Wq, Wqb, HIDDEN * HIDDEN);
  cast_f32_bf16<<<1024, 256, 0, stream>>>(Wk, Wkb, KVHEADS * HDIM * HIDDEN);
  cast_f32_bf16<<<1024, 256, 0, stream>>>(Wv, Wvb, KVHEADS * HDIM * HIDDEN);
  cast_f32_bf16<<<2048, 256, 0, stream>>>(Wo, Wob, HIDDEN * HIDDEN);

  // QKV projections (bf16 WMMA), head-major bf16 outputs
  gemm_bf16_kernel<<<dim3(HIDDEN / 128, MTOT / 128), 256, 0, stream>>>(
      hbf, Wqb, Qh, HIDDEN, 0, NHEADS);
  gemm_bf16_kernel<<<dim3((KVHEADS * HDIM) / 128, MTOT / 128), 256, 0, stream>>>(
      hbf, Wkb, Kh, KVHEADS * HDIM, 0, KVHEADS);
  gemm_bf16_kernel<<<dim3((KVHEADS * HDIM) / 128, MTOT / 128), 256, 0, stream>>>(
      hbf, Wvb, Vh, KVHEADS * HDIM, 0, KVHEADS);

  rope_kernel<<<NHEADS * 1024, 256, 0, stream>>>(Qh, pos, NHEADS);
  rope_kernel<<<KVHEADS * 1024, 256, 0, stream>>>(Kh, pos, KVHEADS);

  attn_kernel<<<BATCH * NHEADS * (SLEN / 64), 128, 0, stream>>>(Qh, Kh, Vh, AO);

  // output projection -> fp32 d_out [B,S,4096]
  gemm_bf16_kernel<<<dim3(HIDDEN / 128, MTOT / 128), 256, 0, stream>>>(
      AO, Wob, d_out, HIDDEN, 1, 0);
}
